// MultiHeadSelfAttentionBlock_22402549416480
// MI455X (gfx1250) — matrix operand order repaired and checked by reading, not compiled, not run on hardware
//
#include <hip/hip_runtime.h>
#include <hip/hip_bf16.h>

// Problem constants (from reference): N=2, T=2048, D=1024, H=16, D_K=64
constexpr int kN  = 2;
constexpr int kT  = 2048;
constexpr int kD  = 1024;
constexpr int kH  = 16;
constexpr int kDK = 64;
constexpr int k3D = 3 * kD;

typedef __attribute__((ext_vector_type(16))) _Float16 v16h;
typedef __attribute__((ext_vector_type(8)))  _Float16 v8h;
typedef __attribute__((ext_vector_type(8)))  float    v8f;
typedef __attribute__((ext_vector_type(4)))  float    v4f;

// ---------------------------------------------------------------------------
// WMMA helper: D = A(16x32 f16) * B(32x16 f16) + C (f32)
// ---------------------------------------------------------------------------
__device__ __forceinline__ v8f wmma_f32_f16(v16h a, v16h b, v8f c) {
  return __builtin_amdgcn_wmma_f32_16x16x32_f16(
      /*neg_a=*/false, a, /*neg_b=*/false, b,
      /*c_mod=*/(short)0, c, /*reuse_a=*/false, /*reuse_b=*/false);
}

// A-matrix fragment (16x32, 16-bit), contiguous-k layout (same k order as load_b_f16):
//   lane L: row M = L%16; elems 0..15 -> K = (L/16)*16 + 0..15
__device__ __forceinline__ v16h load_a_f16(const _Float16* base, int ld, int lane) {
  const int row = lane & 15;
  const int hi  = lane >> 4;
  const _Float16* p = base + (size_t)row * ld + hi * 16;
  const v8h lo = *(const v8h*)(p);
  const v8h hh = *(const v8h*)(p + 8);
  v16h a;
#pragma unroll
  for (int i = 0; i < 8; ++i) { a[i] = lo[i]; a[i + 8] = hh[i]; }
  return a;
}

// Same A fragment sourced from fp32 (global z or LDS probs), b128 loads + cvt.
__device__ __forceinline__ v16h load_a_f32cvt(const float* base, int ld, int lane) {
  const int row = lane & 15;
  const int hi  = lane >> 4;
  const float* p = base + (size_t)row * ld + hi * 16;
  const v4f x0 = *(const v4f*)(p);
  const v4f x1 = *(const v4f*)(p + 4);
  const v4f y0 = *(const v4f*)(p + 8);
  const v4f y1 = *(const v4f*)(p + 12);
  v16h a;
#pragma unroll
  for (int i = 0; i < 4; ++i) {
    a[i]      = (_Float16)x0[i];
    a[i + 4]  = (_Float16)x1[i];
    a[i + 8]  = (_Float16)y0[i];
    a[i + 12] = (_Float16)y1[i];
  }
  return a;
}

// B-matrix fragment (32x16, 16-bit): lane L: col N = L%16;
//   elems e -> K = (L/16)*16 + e  => one contiguous 32B load per lane.
// `base` points at a row-major [N][K]-transposed store: base + n*ld + k.
__device__ __forceinline__ v16h load_b_f16(const _Float16* base, int ld, int lane) {
  const int row = lane & 15;
  const int hi  = lane >> 4;
  return *(const v16h*)(base + (size_t)row * ld + hi * 16);
}

// ---------------------------------------------------------------------------
// Kernel 0: convert fp32 weight [R][C] to f16 transposed [C][R]
// ---------------------------------------------------------------------------
__global__ void cvt_transpose_kernel(const float* __restrict__ src,
                                     _Float16* __restrict__ dst,
                                     int R, int C) {
  size_t idx = (size_t)blockIdx.x * blockDim.x + threadIdx.x;
  if (idx >= (size_t)R * C) return;
  int r = (int)(idx / C);
  int c = (int)(idx % C);
  dst[(size_t)c * R + r] = (_Float16)src[idx];
}

// ---------------------------------------------------------------------------
// Kernel 1: qkv = z @ Wqkv  (f32 in, f16 out, f32 accumulate via WMMA)
// grid = (N*T/16, 3D/256), block = 128 (4 waves; wave w -> 16x64 strip)
// 4x N-blocking: one A fragment feeds 4 WMMAs -> 3 b128 loads per WMMA.
// Writes Q (scaled by dk^-1/2) [N,H,T,64], K [N,H,T,64], V^T [N,H,64,T]
// ---------------------------------------------------------------------------
__global__ void qkv_proj_kernel(const float* __restrict__ z,
                                const _Float16* __restrict__ WqkvT,
                                _Float16* __restrict__ Qh,
                                _Float16* __restrict__ Kh,
                                _Float16* __restrict__ VT) {
  const int tid  = threadIdx.x;
  const int lane = tid & 31;
  const int wave = tid >> 5;
  const int rt   = blockIdx.x;                    // 16-row tile over N*T
  const int c0   = blockIdx.y * 256 + wave * 64;  // 64-col strip in [0,3072)

  const float*    abase = z + (size_t)rt * 16 * kD;
  const _Float16* bbase = WqkvT + (size_t)c0 * kD;

  v8f acc[4] = {{}, {}, {}, {}};
  for (int k = 0; k < kD; k += 32) {
    if (k + 256 < kD) {
      __builtin_prefetch(bbase + (size_t)lane * kD + k + 256, 0, 1);
      __builtin_prefetch(bbase + (size_t)(lane + 32) * kD + k + 256, 0, 1);
    }
    const v16h a = load_a_f32cvt(abase + k, kD, lane);
#pragma unroll
    for (int f = 0; f < 4; ++f) {
      const v16h b = load_b_f16(bbase + (size_t)(f * 16) * kD + k, kD, lane);
      acc[f] = wmma_f32_f16(a, b, acc[f]);
    }
  }

  // Epilogue: scatter four 16x16 tiles into Q / K / V^T (f16).
  const int m0 = (lane >> 4) * 8;
  const float qscale = 0.125f;       // dk^-0.5, dk = 64
#pragma unroll
  for (int f = 0; f < 4; ++f) {
    const int col   = c0 + f * 16 + (lane & 15);
    const int which = col >> 10;       // 0=Q 1=K 2=V
    const int rem   = col & 1023;
    const int head  = rem >> 6;
    const int feat  = rem & 63;
#pragma unroll
    for (int r = 0; r < 8; ++r) {
      const int grow = rt * 16 + m0 + r;     // row in [0, N*T)
      const int n = grow >> 11;              // / T
      const int t = grow & (kT - 1);
      const float v = acc[f][r];
      if (which == 0) {
        Qh[(((size_t)(n * kH + head)) * kT + t) * kDK + feat] = (_Float16)(v * qscale);
      } else if (which == 1) {
        Kh[(((size_t)(n * kH + head)) * kT + t) * kDK + feat] = (_Float16)v;
      } else {
        VT[(((size_t)(n * kH + head)) * kDK + feat) * kT + t] = (_Float16)v;
      }
    }
  }
}

// ---------------------------------------------------------------------------
// Kernel 2: attention. grid = N*T/16 blocks, block = 256 (8 waves).
// Each block owns a 16-query-row tile, loops all 16 heads:
//   scores = Qs @ K^T + mask -> LDS fp32 [16][T]   (waves split key tiles)
//   softmax: vectorized v4f sweeps, 16 threads per row
//   attn-mean accumulated race-free into d_out attn region (v4f rmw)
//   ctx = P @ V -> f16 ctx[N,T,D]  (waves split key dim, LDS reduce)
// Dynamic LDS: 16*T f32 scores + 8*16*64 f32 ctx partials + reductions.
// ---------------------------------------------------------------------------
__global__ void attn_kernel(const _Float16* __restrict__ Qh,
                            const _Float16* __restrict__ Kh,
                            const _Float16* __restrict__ VT,
                            const float* __restrict__ mask,
                            _Float16* __restrict__ ctxbuf,
                            float* __restrict__ attn_out) {
  extern __shared__ char smem_raw[];
  float* sc     = (float*)smem_raw;          // [16][kT]
  float* red    = sc + 16 * kT;              // [8][16][64] ctx partials
  float* rowred = red + 8 * 16 * 64;         // [16][16]
  float* rowmax = rowred + 16 * 16;          // [16]
  float* rowinv = rowmax + 16;               // [16]

  const int tid  = threadIdx.x;
  const int lane = tid & 31;
  const int wave = tid >> 5;                 // 0..7
  const int rt   = blockIdx.x;
  const int n    = rt / (kT / 16);
  const int t0   = (rt % (kT / 16)) * 16;

  const int seg = tid & 15;   // softmax: 16 threads per row
  const int row = tid >> 4;   // 0..15

  for (int h = 0; h < kH; ++h) {
    const _Float16* Qbase = Qh + (((size_t)(n * kH + h)) * kT + t0) * kDK;
    const _Float16* Kbase = Kh + ((size_t)(n * kH + h)) * kT * kDK;
    const _Float16* Vbase = VT + ((size_t)(n * kH + h)) * kDK * kT;

    // Q fragments for k=0..31 and k=32..63 (fixed for whole head)
    const v16h aq0 = load_a_f16(Qbase + 0,  kDK, lane);
    const v16h aq1 = load_a_f16(Qbase + 32, kDK, lane);

    // ---- scores: 8 waves split the 128 key tiles ----
    for (int kt = wave; kt < kT / 16; kt += 8) {
      v8f acc = {};
      v16h b0 = load_b_f16(Kbase + (size_t)kt * 16 * kDK + 0,  kDK, lane);
      acc = wmma_f32_f16(aq0, b0, acc);
      v16h b1 = load_b_f16(Kbase + (size_t)kt * 16 * kDK + 32, kDK, lane);
      acc = wmma_f32_f16(aq1, b1, acc);

      const int col = kt * 16 + (lane & 15);
      const float mv = mask[n * kT + col];
      const int m0 = (lane >> 4) * 8;
#pragma unroll
      for (int r = 0; r < 8; ++r)
        sc[(size_t)(m0 + r) * kT + col] = acc[r] + mv;
    }
    __syncthreads();

    // ---- softmax stats (16 threads per row, 128 cols each, v4f sweeps) ----
    {
      float* srow = sc + (size_t)row * kT;
      const int cbase = seg * 128;

      v4f m4 = *(const v4f*)(srow + cbase);
      for (int c = cbase + 4; c < cbase + 128; c += 4) {
        const v4f x = *(const v4f*)(srow + c);
#pragma unroll
        for (int j = 0; j < 4; ++j) m4[j] = fmaxf(m4[j], x[j]);
      }
      rowred[row * 16 + seg] = fmaxf(fmaxf(m4[0], m4[1]), fmaxf(m4[2], m4[3]));
      __syncthreads();
      if (seg == 0) {
        float mm = rowred[row * 16];
#pragma unroll
        for (int i = 1; i < 16; ++i) mm = fmaxf(mm, rowred[row * 16 + i]);
        rowmax[row] = mm;
      }
      __syncthreads();
      const float mm = rowmax[row];
      float s = 0.f;
      for (int c = cbase; c < cbase + 128; c += 4) {
        v4f x = *(const v4f*)(srow + c);
        v4f e;
#pragma unroll
        for (int j = 0; j < 4; ++j) e[j] = __expf(x[j] - mm);
        s += (e[0] + e[1]) + (e[2] + e[3]);
        *(v4f*)(srow + c) = e;      // keep unnormalized exp
      }
      rowred[row * 16 + seg] = s;
      __syncthreads();
      if (seg == 0) {
        float ss = 0.f;
#pragma unroll
        for (int i = 0; i < 16; ++i) ss += rowred[row * 16 + i];
        rowinv[row] = 1.0f / ss;
      }
      __syncthreads();
    }

    // ---- head-mean of attn, race-free v4f rmw (block owns its rows) ----
    {
      float* aout = attn_out + ((size_t)(n * kT + t0)) * kT;
      const float invH = 1.0f / (float)kH;
      for (int i4 = tid; i4 < 16 * kT / 4; i4 += 256) {
        const int i  = i4 * 4;
        const int rr = i >> 11;                 // / kT (row constant in v4)
        const float w = rowinv[rr] * invH;
        const v4f p = *(const v4f*)(sc + i);
        v4f val;
#pragma unroll
        for (int j = 0; j < 4; ++j) val[j] = p[j] * w;
        if (h == 0) {
          *(v4f*)(aout + i) = val;
        } else {
          v4f old = *(const v4f*)(aout + i);
#pragma unroll
          for (int j = 0; j < 4; ++j) old[j] += val[j];
          *(v4f*)(aout + i) = old;
        }
      }
    }

    // ---- ctx = P @ V : 8 waves split key dim, 4 feat tiles each ----
    {
      v8f acc[4] = {{}, {}, {}, {}};
      for (int kk = wave; kk < kT / 32; kk += 8) {
        const v16h ap = load_a_f32cvt(sc + kk * 32, kT, lane);
#pragma unroll
        for (int f = 0; f < 4; ++f) {
          const v16h bv = load_b_f16(Vbase + (size_t)(f * 16) * kT + kk * 32, kT, lane);
          acc[f] = wmma_f32_f16(ap, bv, acc[f]);
        }
      }
      const int m0 = (lane >> 4) * 8;
      const int cc = lane & 15;
#pragma unroll
      for (int f = 0; f < 4; ++f)
#pragma unroll
        for (int r = 0; r < 8; ++r)
          red[(size_t)(wave * 16 + m0 + r) * 64 + f * 16 + cc] = acc[f][r];
    }
    __syncthreads();

    // ---- reduce partials, normalize, store ctx (f16) ----
    for (int i = tid; i < 16 * 64; i += 256) {
      const int rr = i >> 6, cc = i & 63;
      float v = 0.f;
#pragma unroll
      for (int w = 0; w < 8; ++w)
        v += red[(size_t)(w * 16 + rr) * 64 + cc];
      v *= rowinv[rr];
      ctxbuf[((size_t)(n * kT + t0 + rr)) * kD + h * kDK + cc] = (_Float16)v;
    }
    __syncthreads();  // sc/red reused by next head
  }
}

// ---------------------------------------------------------------------------
// Kernel 3: out = ctx @ Wout  (f16 x f16 -> f32 into d_out)
// grid = (N*T/16, D/256), block = 128 (4 waves; wave w -> 16x64 strip)
// ---------------------------------------------------------------------------
__global__ void out_proj_kernel(const _Float16* __restrict__ ctxbuf,
                                const _Float16* __restrict__ WoutT,
                                float* __restrict__ out) {
  const int tid  = threadIdx.x;
  const int lane = tid & 31;
  const int wave = tid >> 5;
  const int rt   = blockIdx.x;
  const int c0   = blockIdx.y * 256 + wave * 64;

  const _Float16* abase = ctxbuf + (size_t)rt * 16 * kD;
  const _Float16* bbase = WoutT + (size_t)c0 * kD;

  v8f acc[4] = {{}, {}, {}, {}};
  for (int k = 0; k < kD; k += 32) {
    if (k + 256 < kD) {
      __builtin_prefetch(bbase + (size_t)lane * kD + k + 256, 0, 1);
      __builtin_prefetch(bbase + (size_t)(lane + 32) * kD + k + 256, 0, 1);
    }
    const v16h a = load_a_f16(abase + k, kD, lane);
#pragma unroll
    for (int f = 0; f < 4; ++f) {
      const v16h b = load_b_f16(bbase + (size_t)(f * 16) * kD + k, kD, lane);
      acc[f] = wmma_f32_f16(a, b, acc[f]);
    }
  }

  const int m0 = (lane >> 4) * 8;
#pragma unroll
  for (int f = 0; f < 4; ++f) {
    const int col = c0 + f * 16 + (lane & 15);
#pragma unroll
    for (int r = 0; r < 8; ++r)
      out[(size_t)(rt * 16 + m0 + r) * kD + col] = acc[f][r];
  }
}

// ---------------------------------------------------------------------------
extern "C" void kernel_launch(void* const* d_in, const int* in_sizes, int n_in,
                              void* d_out, int out_size, void* d_ws, size_t ws_size,
                              hipStream_t stream) {
  const float* z    = (const float*)d_in[0];   // [N,T,D]
  const float* mask = (const float*)d_in[1];   // [N,1,1,T]
  const float* Wqkv = (const float*)d_in[2];   // [D,3D]
  const float* Wout = (const float*)d_in[3];   // [D,D]

  float* out  = (float*)d_out;                         // [N,T,D]
  float* attn = out + (size_t)kN * kT * kD;            // [N,T,T]

  // Workspace carve-up (all f16, all 32B-aligned chunk sizes)
  char* ws = (char*)d_ws;
  _Float16* WqkvT = (_Float16*)ws; ws += (size_t)k3D * kD * 2;   // 6 MB
  _Float16* WoutT = (_Float16*)ws; ws += (size_t)kD * kD * 2;    // 2 MB
  _Float16* Qh    = (_Float16*)ws; ws += (size_t)kN * kH * kT * kDK * 2;  // 8 MB
  _Float16* Kh    = (_Float16*)ws; ws += (size_t)kN * kH * kT * kDK * 2;  // 8 MB
  _Float16* VT    = (_Float16*)ws; ws += (size_t)kN * kH * kDK * kT * 2;  // 8 MB
  _Float16* ctxb  = (_Float16*)ws; ws += (size_t)kN * kT * kD * 2;        // 8 MB

  // 0) weight convert + transpose
  {
    size_t tot = (size_t)kD * k3D;
    cvt_transpose_kernel<<<dim3((unsigned)((tot + 255) / 256)), 256, 0, stream>>>(
        Wqkv, WqkvT, kD, k3D);
    size_t tot2 = (size_t)kD * kD;
    cvt_transpose_kernel<<<dim3((unsigned)((tot2 + 255) / 256)), 256, 0, stream>>>(
        Wout, WoutT, kD, kD);
  }

  // 1) QKV projection (4x N-blocked WMMA)
  qkv_proj_kernel<<<dim3(kN * kT / 16, k3D / 256), 128, 0, stream>>>(
      z, WqkvT, Qh, Kh, VT);

  // 2) attention (dynamic LDS: scores + ctx partials + reductions)
  {
    size_t smem = (size_t)(16 * kT + 8 * 16 * 64 + 16 * 16 + 16 + 16) * sizeof(float);
    attn_kernel<<<dim3(kN * kT / 16), 256, smem, stream>>>(
        Qh, Kh, VT, mask, ctxb, attn);
  }

  // 3) output projection (4x N-blocked WMMA)
  out_proj_kernel<<<dim3(kN * kT / 16, kD / 256), 128, 0, stream>>>(
      ctxb, WoutT, out);
}
